// MultiHeadAttention_53480932770405
// MI455X (gfx1250) — matrix operand order repaired and checked by reading, not compiled, not run on hardware
//
#include <hip/hip_runtime.h>

// MI455X / gfx1250, wave32. WMMA bf16 + async-LDS / TDM staging.
typedef __bf16 bf16;
typedef __attribute__((ext_vector_type(8)))  bf16  v8bf;
typedef __attribute__((ext_vector_type(16))) bf16  v16bf;
typedef __attribute__((ext_vector_type(8)))  float v8f;
typedef unsigned int u32x4 __attribute__((ext_vector_type(4)));
typedef int          i32x8 __attribute__((ext_vector_type(8)));
typedef int          i32x4 __attribute__((ext_vector_type(4)));
typedef int          v4i   __attribute__((ext_vector_type(4)));

typedef __attribute__((address_space(1))) v4i*  as1_v4i_p;
typedef __attribute__((address_space(3))) v4i*  as3_v4i_p;
typedef __attribute__((address_space(3))) char* as3_char_p;

#define S_LEN    4096
#define D_MODEL  2048
#define N_HEADS  16
#define HEAD_DIM 128

#define NEG_INF (-__builtin_inff())

// ---------------------------------------------------------------------------
// gfx1250 async data movement helpers
// ---------------------------------------------------------------------------
static __device__ __forceinline__ void async_copy_b128(const void* g, void* l) {
#if __has_builtin(__builtin_amdgcn_global_load_async_to_lds_b128)
  __builtin_amdgcn_global_load_async_to_lds_b128((as1_v4i_p)(v4i*)g,
                                                 (as3_v4i_p)(v4i*)l, 0, 0);
#else
  *(v8bf*)l = *(const v8bf*)g;  // fallback: global_load + ds_store
#endif
}

static __device__ __forceinline__ void wait_async() {
#if __has_builtin(__builtin_amdgcn_s_wait_asynccnt)
  __builtin_amdgcn_s_wait_asynccnt(0);
#else
  asm volatile("s_wait_asynccnt 0x0" ::: "memory");
#endif
}

static __device__ __forceinline__ void wait_tensor() {
#if __has_builtin(__builtin_amdgcn_s_wait_tensorcnt)
  __builtin_amdgcn_s_wait_tensorcnt(0);
#else
  asm volatile("s_wait_tensorcnt 0x0" ::: "memory");
#endif
}

// TDM: DMA one contiguous 32 KB tile (4096 x 8B elements) Global -> LDS.
// D# per CDNA5 ISA ch.8: group0 = {count|flags, lds_addr, gaddr_lo,
// gaddr_hi|type=2}; group1 packs data_size=3 (8B), tensor_dim0 = tile_dim0 =
// 4096, tensor_dim1 = tile_dim1 = 1, dim0_stride = 4096. Groups 2/3 unused.
// 6-arg builtin form: (g0, g1, g2, g3, extra, cpol).
static __device__ __forceinline__ void tdm_load_32k(const void* g, void* l) {
#if __has_builtin(__builtin_amdgcn_tensor_load_to_lds)
  unsigned long long ga = (unsigned long long)g;
  unsigned int la = (unsigned int)(unsigned long long)(as3_char_p)l;
  u32x4 g0 = {1u, la, (unsigned int)ga,
              (unsigned int)((ga >> 32) & 0x01FFFFFFu) | (2u << 30)};
  i32x8 g1 = {3 << 16, 4096 << 16, 1 << 16, 4096 << 16, 1, 4096, 0, 0};
  i32x4 z4 = {0, 0, 0, 0};
  i32x8 z8 = {0, 0, 0, 0, 0, 0, 0, 0};
  __builtin_amdgcn_tensor_load_to_lds(g0, g1, z4, z4, z8, 0);
#else
  // fallback: cooperative wave copy (one wave, 32 lanes x 64 b128 chunks)
  const int lane = threadIdx.x & 31;
  for (int i = lane; i < 2048; i += 32)
    async_copy_b128((const bf16*)g + i * 8, (bf16*)l + i * 8);
#endif
}

// ---------------------------------------------------------------------------
// WMMA fragment loaders (layouts per CDNA5 ISA 7.12.2, wave32)
// A 16x32 bf16: lane (m=lane&15, h=lane>>4); two contiguous 8-elem runs.
// B 32x16 bf16: lane (n=lane&15, h=lane>>4); two contiguous 8-elem runs from
//   row n of a row-major K-contiguous matrix.
// C/D 16x16 f32: VGPR r, lane: element (m = r + 8*(lane>>4), n = lane&15).
// ---------------------------------------------------------------------------
static __device__ __forceinline__ v16bf load_a_frag(const bf16* p, int ld) {
  const int lane = threadIdx.x & 31;
  const int m = lane & 15, h = lane >> 4;
  const v8bf a0 = *(const v8bf*)(p + (size_t)m * ld + h * 8);
  const v8bf a1 = *(const v8bf*)(p + (size_t)m * ld + 16 + h * 8);
  v16bf r;
#pragma unroll
  for (int i = 0; i < 8; ++i) { r[i] = a0[i]; r[i + 8] = a1[i]; }
  return r;
}

static __device__ __forceinline__ v16bf load_b_frag(const bf16* p, int ld) {
  const int lane = threadIdx.x & 31;
  const int n = lane & 15, h = lane >> 4;
  const v8bf b0 = *(const v8bf*)(p + (size_t)n * ld + h * 8);
  const v8bf b1 = *(const v8bf*)(p + (size_t)n * ld + 16 + h * 8);
  v16bf r;
#pragma unroll
  for (int i = 0; i < 8; ++i) { r[i] = b0[i]; r[i + 8] = b1[i]; }
  return r;
}

static __device__ __forceinline__ v8f wmma_bf16(v16bf a, v16bf b, v8f c) {
  return __builtin_amdgcn_wmma_f32_16x16x32_bf16(false, a, false, b,
                                                 (short)0, c, false, false);
}

// ---------------------------------------------------------------------------
// fp32 -> bf16 conversion
// ---------------------------------------------------------------------------
__global__ void cvt_to_bf16(const float* __restrict__ in,
                            bf16* __restrict__ out, int n) {
  int i = blockIdx.x * blockDim.x + threadIdx.x;
  if (i < n) out[i] = (bf16)in[i];
}

// ---------------------------------------------------------------------------
// Tiled GEMM: C(MxN) = A(MxK) * W^T, W stored (N x K) row-major.
// Block = 128x128 tile, 8 waves, each wave a 16(M) x 128(N) strip.
// A/B tiles async-staged into LDS (64 KB) per 128-deep K stage; each tile is
// fetched once per block instead of once per wave.
// MODE 1: bf16 out, fused RoPE, layout (head, s, dh)      (Q, K)
// MODE 2: bf16 out, transposed layout (head, dh, s)       (V)
// MODE 3: f32 out, plain (s, d) layout                    (final proj)
// ---------------------------------------------------------------------------
enum { MODE_ROPE = 1, MODE_VT = 2, MODE_F32 = 3 };

template <int MODE>
__global__ __launch_bounds__(256) void proj_gemm(const bf16* __restrict__ A,
                                                 const bf16* __restrict__ W,
                                                 void* __restrict__ outp) {
  __shared__ bf16 As[128 * 128];  // 32 KB
  __shared__ bf16 Bs[128 * 128];  // 32 KB

  const int K    = D_MODEL;
  const int wv   = threadIdx.x >> 5;
  const int lane = threadIdx.x & 31;
  const int nn   = lane & 15, hh = lane >> 4;
  const int rowb = blockIdx.y * 128;
  const int colb = blockIdx.x * 128;
  const int row0 = rowb + wv * 16;

  v8f acc[8] = {};

  for (int ks = 0; ks < K; ks += 128) {
    // cooperative async stage: 2048 b128 chunks per tile, 8+8 per thread
    for (int i = threadIdx.x; i < 2048; i += 256) {
      const int r = i >> 4, seg = (i & 15) * 8;
      async_copy_b128(A + (size_t)(rowb + r) * K + ks + seg,
                      As + r * 128 + seg);
      async_copy_b128(W + (size_t)(colb + r) * K + ks + seg,
                      Bs + r * 128 + seg);
    }
    wait_async();
    __syncthreads();

#pragma unroll
    for (int kk = 0; kk < 128; kk += 32) {
      v16bf a = load_a_frag(As + (wv * 16) * 128 + kk, 128);
#pragma unroll
      for (int nt = 0; nt < 8; ++nt) {
        v16bf b = load_b_frag(Bs + (nt * 16) * 128 + kk, 128);
        acc[nt] = wmma_bf16(a, b, acc[nt]);
      }
    }
    __syncthreads();
  }

  if (MODE == MODE_ROPE) {
    // NeoX rotation: pair (i, i+64); wave owns the full head width so both
    // halves are in-register (tiles t and t+4).
#pragma unroll
    for (int t = 0; t < 4; ++t) {
#pragma unroll
      for (int r = 0; r < 8; ++r) {
        const int   i     = t * 16 + nn;  // 0..63
        const float inv_f = __powf(10000.0f, -(float)i * (1.0f / 64.0f));
        const float ang   = (float)(row0 + r + 8 * hh) * inv_f;
        float sn, cs;
        __sincosf(ang, &sn, &cs);
        const float x1 = acc[t][r], x2 = acc[t + 4][r];
        acc[t][r]     = x1 * cs - x2 * sn;
        acc[t + 4][r] = x2 * cs + x1 * sn;
      }
    }
  }

  if (MODE == MODE_F32) {
    float* out = (float*)outp;
#pragma unroll
    for (int nt = 0; nt < 8; ++nt)
#pragma unroll
      for (int r = 0; r < 8; ++r)
        out[(size_t)(row0 + r + 8 * hh) * D_MODEL + colb + nt * 16 + nn] =
            acc[nt][r];
  } else if (MODE == MODE_VT) {
    bf16* out = (bf16*)outp + (size_t)blockIdx.x * HEAD_DIM * S_LEN;
#pragma unroll
    for (int nt = 0; nt < 8; ++nt)
#pragma unroll
      for (int r = 0; r < 8; ++r)
        out[(size_t)(nt * 16 + nn) * S_LEN + row0 + r + 8 * hh] =
            (bf16)acc[nt][r];
  } else {
    bf16* out = (bf16*)outp + (size_t)blockIdx.x * S_LEN * HEAD_DIM;
#pragma unroll
    for (int nt = 0; nt < 8; ++nt)
#pragma unroll
      for (int r = 0; r < 8; ++r)
        out[(size_t)(row0 + r + 8 * hh) * HEAD_DIM + nt * 16 + nn] =
            (bf16)acc[nt][r];
  }
}

// ---------------------------------------------------------------------------
// Flash attention, one block per (head, 128-query tile), 8 waves.
// Per KV block: the contiguous 32 KB K tile comes in via TDM
// (tensor_load_to_lds, wave 0), the strided V^T tile via async b128 copies.
// All waves consume WMMA B-fragments from LDS. P staged through per-wave LDS
// to reshape D-layout -> A-layout. 96 KB LDS total (320 KB WGP budget).
// ---------------------------------------------------------------------------
__global__ __launch_bounds__(256) void flash_attn(const bf16* __restrict__ Q,
                                                  const bf16* __restrict__ Kb,
                                                  const bf16* __restrict__ Vt,
                                                  bf16* __restrict__ ctx) {
  __shared__ bf16 Ks[128 * 128];        // 32 KB  [kv_local][dh]
  __shared__ bf16 Vs[128 * 128];        // 32 KB  [dh][kv_local]
  __shared__ bf16 plds[8 * 16 * 128];   // 32 KB  per-wave P strips

  const int head = blockIdx.x;
  const int qb   = blockIdx.y;
  const int wv   = threadIdx.x >> 5;
  const int lane = threadIdx.x & 31;
  const int nn   = lane & 15, hh = lane >> 4;
  const int q0   = qb * 128 + wv * 16;

  const bf16* Qh = Q  + (size_t)head * S_LEN * HEAD_DIM;
  const bf16* Kh = Kb + (size_t)head * S_LEN * HEAD_DIM;
  const bf16* Vh = Vt + (size_t)head * HEAD_DIM * S_LEN;

  v16bf aq[4];
#pragma unroll
  for (int kc = 0; kc < 4; ++kc)
    aq[kc] = load_a_frag(Qh + (size_t)q0 * HEAD_DIM + kc * 32, HEAD_DIM);

  v8f   o[8] = {};
  float mrow[8], lrow[8];
#pragma unroll
  for (int r = 0; r < 8; ++r) { mrow[r] = NEG_INF; lrow[r] = 0.0f; }

  const float scale = 0.08838834764831845f;  // 1/sqrt(128)
  bf16* pw = plds + wv * 16 * HEAD_DIM;

  for (int jb = 0; jb <= qb; ++jb) {
    // ---- stage K (TDM) and V^T (async copies) tiles into LDS ----
    if (wv == 0)
      tdm_load_32k(Kh + (size_t)jb * 128 * HEAD_DIM, Ks);
    for (int i = threadIdx.x; i < 2048; i += 256) {
      const int r = i >> 4, seg = (i & 15) * 8;
      async_copy_b128(Vh + (size_t)r * S_LEN + jb * 128 + seg,
                      Vs + r * 128 + seg);
    }
    wait_async();
    wait_tensor();
    __syncthreads();

    // ---- S tile: 16 x 128 scores vs this KV block ----
    v8f sfr[8];
#pragma unroll
    for (int nt = 0; nt < 8; ++nt) {
      v8f sa = {};
#pragma unroll
      for (int kc = 0; kc < 4; ++kc) {
        v16bf b = load_b_frag(Ks + (nt * 16) * HEAD_DIM + kc * 32, HEAD_DIM);
        sa = wmma_bf16(aq[kc], b, sa);
      }
      sfr[nt] = sa * scale;
    }

    if (jb == qb) {  // causal mask only on the diagonal block
#pragma unroll
      for (int nt = 0; nt < 8; ++nt)
#pragma unroll
        for (int r = 0; r < 8; ++r) {
          const int kv = jb * 128 + nt * 16 + nn;
          const int qr = q0 + r + 8 * hh;
          if (kv > qr) sfr[nt][r] = NEG_INF;
        }
    }

    // ---- online softmax (rows live across 16 lanes of one VGPR) ----
    float alpha[8];
#pragma unroll
    for (int r = 0; r < 8; ++r) {
      float tm = sfr[0][r];
#pragma unroll
      for (int nt = 1; nt < 8; ++nt) tm = fmaxf(tm, sfr[nt][r]);
#pragma unroll
      for (int off = 8; off >= 1; off >>= 1)
        tm = fmaxf(tm, __shfl_xor(tm, off, 32));
      const float mn = fmaxf(mrow[r], tm);
      alpha[r] = __expf(mrow[r] - mn);
      mrow[r]  = mn;
      float rs = 0.0f;
#pragma unroll
      for (int nt = 0; nt < 8; ++nt) {
        const float p = __expf(sfr[nt][r] - mn);
        sfr[nt][r] = p;
        rs += p;
      }
#pragma unroll
      for (int off = 8; off >= 1; off >>= 1) rs += __shfl_xor(rs, off, 32);
      lrow[r] = lrow[r] * alpha[r] + rs;
    }

    // ---- rescale O, stage P(bf16) into per-wave LDS strip ----
#pragma unroll
    for (int dt = 0; dt < 8; ++dt)
#pragma unroll
      for (int r = 0; r < 8; ++r) o[dt][r] *= alpha[r];
#pragma unroll
    for (int nt = 0; nt < 8; ++nt)
#pragma unroll
      for (int r = 0; r < 8; ++r)
        pw[(r + 8 * hh) * HEAD_DIM + nt * 16 + nn] = (bf16)sfr[nt][r];

    // ---- O(16x128) += P(16x128) * V(128x128), V^T fragments from LDS ----
#pragma unroll
    for (int kc = 0; kc < 4; ++kc) {
      v16bf ap = load_a_frag(pw + kc * 32, HEAD_DIM);
#pragma unroll
      for (int dt = 0; dt < 8; ++dt) {
        v16bf b = load_b_frag(Vs + (dt * 16) * 128 + kc * 32, 128);
        o[dt] = wmma_bf16(ap, b, o[dt]);
      }
    }
    __syncthreads();  // all waves done with Ks/Vs before next stage
  }

  // ---- normalize and write ctx in (s, d_model) layout ----
#pragma unroll
  for (int r = 0; r < 8; ++r) {
    const float inv_l = 1.0f / lrow[r];
    const int   srow  = q0 + r + 8 * hh;
#pragma unroll
    for (int dt = 0; dt < 8; ++dt)
      ctx[(size_t)srow * D_MODEL + head * HEAD_DIM + dt * 16 + nn] =
          (bf16)(o[dt][r] * inv_l);
  }
}

// ---------------------------------------------------------------------------
// Host side
// ---------------------------------------------------------------------------
extern "C" void kernel_launch(void* const* d_in, const int* in_sizes, int n_in,
                              void* d_out, int out_size, void* d_ws,
                              size_t ws_size, hipStream_t stream) {
  (void)in_sizes; (void)n_in; (void)out_size; (void)ws_size;
  const float* x  = (const float*)d_in[0];
  const float* wq = (const float*)d_in[1];
  const float* wk = (const float*)d_in[2];
  const float* wv = (const float*)d_in[3];
  const float* wo = (const float*)d_in[4];

  const size_t SD = (size_t)S_LEN * D_MODEL;    // 8388608
  const size_t WW = (size_t)D_MODEL * D_MODEL;  // 4194304

  bf16* xb   = (bf16*)d_ws;
  bf16* wqb  = xb + SD;
  bf16* wkb  = wqb + WW;
  bf16* wvb  = wkb + WW;
  bf16* wob  = wvb + WW;
  bf16* Qb   = wob + WW;   // (head, s, dh), RoPE applied
  bf16* Kbuf = Qb + SD;    // (head, s, dh), RoPE applied
  bf16* Vtb  = Kbuf + SD;  // (head, dh, s)  transposed
  bf16* ctxb = Vtb + SD;   // (s, d_model)
  // total workspace: ~112 MB

  const int T = 256;
  cvt_to_bf16<<<(int)((SD + T - 1) / T), T, 0, stream>>>(x, xb, (int)SD);
  cvt_to_bf16<<<(int)((WW + T - 1) / T), T, 0, stream>>>(wq, wqb, (int)WW);
  cvt_to_bf16<<<(int)((WW + T - 1) / T), T, 0, stream>>>(wk, wkb, (int)WW);
  cvt_to_bf16<<<(int)((WW + T - 1) / T), T, 0, stream>>>(wv, wvb, (int)WW);
  cvt_to_bf16<<<(int)((WW + T - 1) / T), T, 0, stream>>>(wo, wob, (int)WW);

  dim3 g(D_MODEL / 128, S_LEN / 128);  // 16 x 32 tiles
  proj_gemm<MODE_ROPE><<<g, 256, 0, stream>>>(xb, wqb, (void*)Qb);
  proj_gemm<MODE_ROPE><<<g, 256, 0, stream>>>(xb, wkb, (void*)Kbuf);
  proj_gemm<MODE_VT><<<g, 256, 0, stream>>>(xb, wvb, (void*)Vtb);

  flash_attn<<<dim3(N_HEADS, S_LEN / 128), 256, 0, stream>>>(Qb, Kbuf, Vtb,
                                                             ctxb);

  proj_gemm<MODE_F32><<<g, 256, 0, stream>>>(ctxb, wob, d_out);
}
